// TransformerEncoderLayer_85882166050854
// MI455X (gfx1250) — compile-verified, matrix operand order checked by reading, not run on hardware
//
#include <hip/hip_runtime.h>

// Problem constants (from the reference)
#define Bb   4
#define Ss   1024
#define Dd   1024
#define Hh   16
#define DHh  64
#define DFFf 4096
#define NEGV (-1e9f)

typedef __bf16 bf16_t;
typedef __attribute__((ext_vector_type(8)))  __bf16 v8bf;
typedef __attribute__((ext_vector_type(16))) __bf16 v16bf;
typedef __attribute__((ext_vector_type(8)))  float  v8f;

// ---------------------------------------------------------------- helpers --
__device__ __forceinline__ bf16_t f2bf(float f) {
  union { float f; unsigned int u; } a; a.f = f;
  unsigned int r = a.u + 0x7FFFu + ((a.u >> 16) & 1u);   // round-to-nearest-even
  union { unsigned short u; bf16_t h; } o; o.u = (unsigned short)(r >> 16);
  return o.h;
}

__device__ __forceinline__ v8f zero8() {
  v8f z;
#pragma unroll
  for (int i = 0; i < 8; ++i) z[i] = 0.0f;
  return z;
}

// Assemble a 16-element bf16 A/B operand fragment from two contiguous
// 8-element chunks (wave32 layout: a lane holds K = {hk*8..+7, 16+hk*8..+7}).
__device__ __forceinline__ v16bf frag16(const bf16_t* p0, const bf16_t* p1) {
  v8bf lo = *(const v8bf*)p0;
  v8bf hi = *(const v8bf*)p1;
  return __builtin_shufflevector(lo, hi, 0, 1, 2, 3, 4, 5, 6, 7,
                                 8, 9, 10, 11, 12, 13, 14, 15);
}

__device__ __forceinline__ v8f wmma_bf16(v16bf a, v16bf b, v8f c) {
  // (neg_a, A, neg_b, B, c_mod, C, reuse_a, reuse_b)
  return __builtin_amdgcn_wmma_f32_16x16x32_bf16(false, a, false, b,
                                                 (short)0, c, false, false);
}

// ---- CDNA5 async global->LDS copy (ASYNCcnt path, no VGPR transit) --------
// Per lane: LDS[vdst] = MEM[vaddr .. +15]. LDS byte offset = low 32 bits of
// the generic shared-memory pointer (ISA: LDS aperture keeps offset in [31:0]).
__device__ __forceinline__ void async_ld16(void* lds, const void* gptr) {
  unsigned l = (unsigned)(unsigned long long)lds;
  unsigned long long g = (unsigned long long)gptr;
  asm volatile("global_load_async_to_lds_b128 %0, %1, off"
               :: "v"(l), "v"(g) : "memory");
}
#define WAIT_ASYNC(n) asm volatile("s_wait_asynccnt " #n ::: "memory")

// ----------------------------------------------- transpose-cast fp32->bf16 --
// src[K][N] fp32 -> dst[N][K] bf16 (one-time; weights only). 32x32 LDS tiles.
__global__ __launch_bounds__(256)
void k_castT(const float* __restrict__ src, bf16_t* __restrict__ dst,
             int K, int N) {
  __shared__ bf16_t tile[32][33];
  int kBase = blockIdx.y * 32, nBase = blockIdx.x * 32;
  int tid = threadIdx.x;
  int r = tid >> 3, c4 = (tid & 7) * 4;
  const float* sp = src + (size_t)(kBase + r) * N + nBase + c4;
#pragma unroll
  for (int j = 0; j < 4; ++j) tile[r][c4 + j] = f2bf(sp[j]);
  __syncthreads();
  bf16_t* dp = dst + (size_t)(nBase + r) * K + kBase + c4;
#pragma unroll
  for (int j = 0; j < 4; ++j) dp[j] = tile[c4 + j][r];
}

// --------------------------------------------------------------- LayerNorm
__global__ __launch_bounds__(256)
void k_ln_in(const float* __restrict__ x, const float* __restrict__ g,
             const float* __restrict__ bb, float* __restrict__ outF,
             bf16_t* __restrict__ outB) {
  int row = blockIdx.x;
  int tid = threadIdx.x;
  __shared__ float2 red[256];
  const float* xr = x + (size_t)row * Dd;
  float v[4];
  float s = 0.f, ss = 0.f;
#pragma unroll
  for (int i = 0; i < 4; ++i) {
    float t = xr[tid + i * 256];
    v[i] = t; s += t; ss += t * t;
  }
  red[tid] = make_float2(s, ss);
  __syncthreads();
  for (int off = 128; off > 0; off >>= 1) {
    if (tid < off) {
      red[tid].x += red[tid + off].x;
      red[tid].y += red[tid + off].y;
    }
    __syncthreads();
  }
  float mu = red[0].x * (1.0f / Dd);
  float var = red[0].y * (1.0f / Dd) - mu * mu;
  float rs = rsqrtf(var + 1e-5f);
#pragma unroll
  for (int i = 0; i < 4; ++i) {
    int c = tid + i * 256;
    float o = (v[i] - mu) * rs * g[c] + bb[c];
    outF[(size_t)row * Dd + c] = o;
    outB[(size_t)row * Dd + c] = f2bf(o);
  }
}

// x2 = xln + LN2(attn_proj);  h = bf16(LN3(x2))
__global__ __launch_bounds__(256)
void k_post_attn(const float* __restrict__ xln, const float* __restrict__ ap,
                 const float* __restrict__ g2, const float* __restrict__ b2,
                 const float* __restrict__ g3, const float* __restrict__ b3,
                 float* __restrict__ x2, bf16_t* __restrict__ hB) {
  int row = blockIdx.x;
  int tid = threadIdx.x;
  __shared__ float2 red[256];
  const float* ar = ap + (size_t)row * Dd;
  const float* xr = xln + (size_t)row * Dd;
  float a[4];
  float s = 0.f, ss = 0.f;
#pragma unroll
  for (int i = 0; i < 4; ++i) {
    float t = ar[tid + i * 256];
    a[i] = t; s += t; ss += t * t;
  }
  red[tid] = make_float2(s, ss);
  __syncthreads();
  for (int off = 128; off > 0; off >>= 1) {
    if (tid < off) {
      red[tid].x += red[tid + off].x;
      red[tid].y += red[tid + off].y;
    }
    __syncthreads();
  }
  float mu1 = red[0].x * (1.0f / Dd);
  float var1 = red[0].y * (1.0f / Dd) - mu1 * mu1;
  float rs1 = rsqrtf(var1 + 1e-5f);
  __syncthreads();  // everyone done reading red[0] before reuse

  float y[4];
  float s2 = 0.f, ss2 = 0.f;
#pragma unroll
  for (int i = 0; i < 4; ++i) {
    int c = tid + i * 256;
    float t = xr[c] + (a[i] - mu1) * rs1 * g2[c] + b2[c];
    y[i] = t;
    x2[(size_t)row * Dd + c] = t;
    s2 += t; ss2 += t * t;
  }
  red[tid] = make_float2(s2, ss2);
  __syncthreads();
  for (int off = 128; off > 0; off >>= 1) {
    if (tid < off) {
      red[tid].x += red[tid + off].x;
      red[tid].y += red[tid + off].y;
    }
    __syncthreads();
  }
  float mu2 = red[0].x * (1.0f / Dd);
  float var2 = red[0].y * (1.0f / Dd) - mu2 * mu2;
  float rs2 = rsqrtf(var2 + 1e-5f);
#pragma unroll
  for (int i = 0; i < 4; ++i) {
    int c = tid + i * 256;
    hB[(size_t)row * Dd + c] = f2bf((y[i] - mu2) * rs2 * g3[c] + b3[c]);
  }
}

// ------------------------------------------------------------- WMMA GEMM --
// C[M,N](f32) = A[M,K](bf16, row-major) * B (bf16, PRE-TRANSPOSED [N][K]).
// Block tile 64x128, BK=64, 8 waves * (2x2) frags, 8 WMMAs per stage.
// Double-buffered LDS fed by async global->LDS loads (6 per wave per tile).
__device__ __forceinline__ void gemm_stage(const bf16_t* A, const bf16_t* BT,
                                           bf16_t* As, bf16_t* Bs,
                                           int mBase, int nBase, int K, int k0,
                                           int tid) {
#pragma unroll
  for (int i = 0; i < 2; ++i) {
    int c = tid * 2 + i;           // 0..511
    int row = c >> 3, kc = (c & 7) * 8;
    async_ld16(&As[row * 64 + kc], &A[(size_t)(mBase + row) * K + k0 + kc]);
  }
#pragma unroll
  for (int i = 0; i < 4; ++i) {
    int c = tid * 4 + i;           // 0..1023
    int n = c >> 3, kc = (c & 7) * 8;
    async_ld16(&Bs[n * 64 + kc], &BT[(size_t)(nBase + n) * K + k0 + kc]);
  }
}

__global__ __launch_bounds__(256)
void k_gemm(const bf16_t* __restrict__ A, const bf16_t* __restrict__ BT,
            const float* __restrict__ bias, const float* __restrict__ residual,
            float* __restrict__ outF, bf16_t* __restrict__ outB,
            int M, int N, int K, int act, int vmode) {
  __shared__ __align__(16) bf16_t As[2][64 * 64];    // [m][k]
  __shared__ __align__(16) bf16_t Bs[2][128 * 64];   // [n][k]
  int tid = threadIdx.x;
  int lane = tid & 31, wv = tid >> 5;
  int m = lane & 15, hk = lane >> 4;
  int wm = (wv & 1) * 32, wn = (wv >> 1) * 32;
  int mBase = blockIdx.y * 64, nBase = blockIdx.x * 128;

  v8f acc[2][2];
#pragma unroll
  for (int i = 0; i < 2; ++i)
#pragma unroll
    for (int j = 0; j < 2; ++j) acc[i][j] = zero8();

  int nk = K >> 6;
  gemm_stage(A, BT, As[0], Bs[0], mBase, nBase, K, 0, tid);  // prologue

  for (int kt = 0; kt < nk; ++kt) {
    int cur = kt & 1;
    if (kt + 1 < nk) {
      gemm_stage(A, BT, As[1 - cur], Bs[1 - cur], mBase, nBase, K,
                 (kt + 1) * 64, tid);
      WAIT_ASYNC(6);   // in-order completion => tile kt's 6 loads landed
    } else {
      WAIT_ASYNC(0);
    }
    __syncthreads();

#pragma unroll
    for (int c32 = 0; c32 < 2; ++c32) {
      const bf16_t* a0p = &As[cur][(wm + m) * 64 + c32 * 32];
      const bf16_t* a1p = &As[cur][(wm + 16 + m) * 64 + c32 * 32];
      const bf16_t* b0p = &Bs[cur][(wn + m) * 64 + c32 * 32];
      const bf16_t* b1p = &Bs[cur][(wn + 16 + m) * 64 + c32 * 32];
      v16bf a0 = frag16(a0p + hk * 8, a0p + 16 + hk * 8);
      v16bf a1 = frag16(a1p + hk * 8, a1p + 16 + hk * 8);
      v16bf b0 = frag16(b0p + hk * 8, b0p + 16 + hk * 8);
      v16bf b1 = frag16(b1p + hk * 8, b1p + 16 + hk * 8);
      acc[0][0] = wmma_bf16(a0, b0, acc[0][0]);
      acc[0][1] = wmma_bf16(a0, b1, acc[0][1]);
      acc[1][0] = wmma_bf16(a1, b0, acc[1][0]);
      acc[1][1] = wmma_bf16(a1, b1, acc[1][1]);
    }
    __syncthreads();  // all waves done reading buf before it is re-filled
  }

  // epilogue: C frag layout = VGPR r, lane: M = r + 8*hk, N = lane&15
#pragma unroll
  for (int fm = 0; fm < 2; ++fm)
#pragma unroll
    for (int fn = 0; fn < 2; ++fn)
#pragma unroll
      for (int r = 0; r < 8; ++r) {
        int row = mBase + wm + fm * 16 + r + 8 * hk;
        int col = nBase + wn + fn * 16 + m;
        float v = acc[fm][fn][r] + bias[col];
        if (act == 1) v = 0.5f * v * (1.0f + erff(v * 0.70710678118f));
        if (residual) v += residual[(size_t)row * N + col];
        if (outF) outF[(size_t)row * N + col] = v;
        if (outB) {
          size_t o;
          if (vmode == 1) {
            // V^T per head: vT[b][h][dh][s]; row = b*S+s, col = h*64+dh
            o = (((size_t)(row >> 10) * Hh + (col >> 6)) * DHh + (col & 63)) *
                    Ss + (row & (Ss - 1));
          } else {
            o = (size_t)row * N + col;
          }
          outB[o] = f2bf(v);
        }
      }
}

// -------------------------------------------------- flash attention (bf16) --
// grid = (S/128, H, B), 256 threads = 8 waves; each wave owns 16 query rows.
// vT is the V matrix pre-transposed per head: vT[b][h][dh][s].
// Double-buffered K/V tiles staged with async global->LDS (2 per wave/tile).
__device__ __forceinline__ void attn_stage(const bf16_t* kB, const bf16_t* vT,
                                           bf16_t* Ks, bf16_t* Vt,
                                           int b, int h, size_t vhead,
                                           int key0, int tid) {
  int idx = tid * 8;
  {
    int kr = idx >> 6, c0 = idx & 63;
    async_ld16(&Ks[kr * 64 + c0],
               &kB[(((size_t)(b * Ss + key0 + kr)) * Hh + h) * DHh + c0]);
  }
  {
    int dh = idx >> 5, kc = idx & 31;
    async_ld16(&Vt[dh * 32 + kc], &vT[vhead + (size_t)dh * Ss + key0 + kc]);
  }
}

__global__ __launch_bounds__(256)
void k_attn(const bf16_t* __restrict__ qB, const bf16_t* __restrict__ kB,
            const bf16_t* __restrict__ vT, bf16_t* __restrict__ aB) {
  __shared__ __align__(16) bf16_t Ks[2][32 * 64];   // [key][dh]
  __shared__ __align__(16) bf16_t Vt[2][64 * 32];   // [dh][key]
  __shared__ __align__(16) bf16_t Pw[8 * 16 * 32];  // per-wave P scratch

  int tid = threadIdx.x;
  int lane = tid & 31, wv = tid >> 5;
  int m = lane & 15, hk = lane >> 4;
  int b = blockIdx.z, h = blockIdx.y;
  int q0 = blockIdx.x * 128 + wv * 16;
  int qmax = q0 + 15;

  // Q fragments for two K-steps (DH = 64 = 2 * 32)
  const bf16_t* qp = qB + (((size_t)(b * Ss + q0 + m)) * Hh + h) * DHh;
  v16bf qa[2];
#pragma unroll
  for (int c = 0; c < 2; ++c)
    qa[c] = frag16(qp + c * 32 + hk * 8, qp + c * 32 + 16 + hk * 8);

  v8f acc[4];
  float mrow[8], lrow[8];
#pragma unroll
  for (int t = 0; t < 4; ++t) acc[t] = zero8();
#pragma unroll
  for (int r = 0; r < 8; ++r) { mrow[r] = -3.0e38f; lrow[r] = 0.0f; }

  const size_t vhead = ((size_t)(b * Hh + h)) * DHh * Ss;  // vT head base

  int ntiles = (blockIdx.x + 1) * 4;  // causal upper bound for this block
  attn_stage(kB, vT, Ks[0], Vt[0], b, h, vhead, 0, tid);  // prologue

  for (int kt = 0; kt < ntiles; ++kt) {
    int key0 = kt * 32;
    int cur = kt & 1;
    if (kt + 1 < ntiles) {
      attn_stage(kB, vT, Ks[1 - cur], Vt[1 - cur], b, h, vhead,
                 (kt + 1) * 32, tid);
      WAIT_ASYNC(2);   // in-order completion => tile kt's 2 loads landed
    } else {
      WAIT_ASYNC(0);
    }
    __syncthreads();

    if (key0 <= qmax) {  // uniform per wave
      // logits = Q * K^T for 32 keys (two 16-key N-subtiles)
      v8f lf0 = zero8(), lf1 = zero8();
#pragma unroll
      for (int c = 0; c < 2; ++c) {
        const bf16_t* r0 = &Ks[cur][m * 64 + c * 32];
        const bf16_t* r1 = &Ks[cur][(16 + m) * 64 + c * 32];
        v16bf kf0 = frag16(r0 + hk * 8, r0 + 16 + hk * 8);
        v16bf kf1 = frag16(r1 + hk * 8, r1 + 16 + hk * 8);
        lf0 = wmma_bf16(qa[c], kf0, lf0);
        lf1 = wmma_bf16(qa[c], kf1, lf1);
      }
      // scale + causal/padding mask + online softmax
#pragma unroll
      for (int r = 0; r < 8; ++r) {
        int qrow = q0 + r + 8 * hk;
        int k0e = key0 + m, k1e = key0 + 16 + m;
        float e0 = lf0[r] * 0.125f;  // 1/sqrt(64)
        float e1 = lf1[r] * 0.125f;
        if (k0e > qrow || k0e >= (Ss - 64)) e0 = NEGV;
        if (k1e > qrow || k1e >= (Ss - 64)) e1 = NEGV;
        float mx = fmaxf(e0, e1);
        mx = fmaxf(mx, __shfl_xor(mx, 1));
        mx = fmaxf(mx, __shfl_xor(mx, 2));
        mx = fmaxf(mx, __shfl_xor(mx, 4));
        mx = fmaxf(mx, __shfl_xor(mx, 8));
        float nm = fmaxf(mrow[r], mx);
        float corr = __expf(mrow[r] - nm);
#pragma unroll
        for (int t = 0; t < 4; ++t) acc[t][r] = acc[t][r] * corr;
        float p0 = __expf(e0 - nm);
        float p1 = __expf(e1 - nm);
        float sl = p0 + p1;
        sl += __shfl_xor(sl, 1);
        sl += __shfl_xor(sl, 2);
        sl += __shfl_xor(sl, 4);
        sl += __shfl_xor(sl, 8);
        lrow[r] = lrow[r] * corr + sl;
        mrow[r] = nm;
        lf0[r] = p0; lf1[r] = p1;
      }
      // re-layout P (C-frag f32) -> A-frag bf16 through per-wave LDS
      bf16_t* P = &Pw[wv * 16 * 32];
#pragma unroll
      for (int r = 0; r < 8; ++r) {
        P[(r + 8 * hk) * 32 + m] = f2bf(lf0[r]);
        P[(r + 8 * hk) * 32 + 16 + m] = f2bf(lf1[r]);
      }
      asm volatile("s_wait_dscnt 0" ::: "memory");  // wave-local DS RAW
      v16bf pa = frag16(&P[m * 32 + hk * 8], &P[m * 32 + 16 + hk * 8]);
      // acc += P(16x32) * V(32x64) as four 16-col WMMAs
#pragma unroll
      for (int t = 0; t < 4; ++t) {
        const bf16_t* vr = &Vt[cur][(t * 16 + m) * 32];
        v16bf vf = frag16(vr + hk * 8, vr + 16 + hk * 8);
        acc[t] = wmma_bf16(pa, vf, acc[t]);
      }
    }
    __syncthreads();
  }

  // write attention output (bf16, [B*S, H*DH] row-major)
#pragma unroll
  for (int t = 0; t < 4; ++t)
#pragma unroll
    for (int r = 0; r < 8; ++r) {
      int q = q0 + r + 8 * hk;
      int col = h * DHh + t * 16 + m;
      float o = acc[t][r] / lrow[r];
      aB[((size_t)(b * Ss + q)) * (Hh * DHh) + col] = f2bf(o);
    }
}

// ------------------------------------------------------------------ launch --
extern "C" void kernel_launch(void* const* d_in, const int* in_sizes, int n_in,
                              void* d_out, int out_size, void* d_ws,
                              size_t ws_size, hipStream_t stream) {
  (void)in_sizes; (void)n_in; (void)out_size; (void)ws_size;

  const float* input = (const float*)d_in[0];
  const float* ln1_g = (const float*)d_in[3];
  const float* ln1_b = (const float*)d_in[4];
  const float* ln2_g = (const float*)d_in[5];
  const float* ln2_b = (const float*)d_in[6];
  const float* ln3_g = (const float*)d_in[7];
  const float* ln3_b = (const float*)d_in[8];
  const float* Wq = (const float*)d_in[9];
  const float* bq = (const float*)d_in[10];
  const float* Wk = (const float*)d_in[11];
  const float* bk = (const float*)d_in[12];
  const float* Wv = (const float*)d_in[13];
  const float* bv = (const float*)d_in[14];
  const float* Wo = (const float*)d_in[15];
  const float* bo = (const float*)d_in[16];
  const float* W1 = (const float*)d_in[17];
  const float* b1 = (const float*)d_in[18];
  const float* W2 = (const float*)d_in[19];
  const float* b2 = (const float*)d_in[20];

  const size_t BS = (size_t)Bb * Ss;  // 4096 rows
  char* p = (char*)d_ws;
  auto bump = [&](size_t bytes) -> char* {
    char* r = p;
    p += (bytes + 255) & ~(size_t)255;
    return r;
  };

  float*  xlnF  = (float*)bump(BS * Dd * 4);
  bf16_t* xlnB  = (bf16_t*)bump(BS * Dd * 2);
  bf16_t* qBf   = (bf16_t*)bump(BS * Dd * 2);
  bf16_t* kBf   = (bf16_t*)bump(BS * Dd * 2);
  bf16_t* vTf   = (bf16_t*)bump(BS * Dd * 2);   // per-head transposed V
  bf16_t* attnB = (bf16_t*)bump(BS * Dd * 2);
  float*  apF   = (float*)bump(BS * Dd * 4);
  float*  x2F   = (float*)bump(BS * Dd * 4);
  bf16_t* hBf   = (bf16_t*)bump(BS * Dd * 2);
  bf16_t* hhBf  = (bf16_t*)bump(BS * DFFf * 2);
  bf16_t* wqT   = (bf16_t*)bump((size_t)Dd * Dd * 2);
  bf16_t* wkT   = (bf16_t*)bump((size_t)Dd * Dd * 2);
  bf16_t* wvT   = (bf16_t*)bump((size_t)Dd * Dd * 2);
  bf16_t* woT   = (bf16_t*)bump((size_t)Dd * Dd * 2);
  bf16_t* w1T   = (bf16_t*)bump((size_t)Dd * DFFf * 2);
  bf16_t* w2T   = (bf16_t*)bump((size_t)DFFf * Dd * 2);

  // 1) LN1
  k_ln_in<<<(int)BS, 256, 0, stream>>>(input, ln1_g, ln1_b, xlnF, xlnB);

  // 2) one-time weight transpose+cast: W[K][N] -> WT[N][K] bf16
  dim3 gT1(Dd / 32, Dd / 32);
  k_castT<<<gT1, 256, 0, stream>>>(Wq, wqT, Dd, Dd);
  k_castT<<<gT1, 256, 0, stream>>>(Wk, wkT, Dd, Dd);
  k_castT<<<gT1, 256, 0, stream>>>(Wv, wvT, Dd, Dd);
  k_castT<<<gT1, 256, 0, stream>>>(Wo, woT, Dd, Dd);
  dim3 gT2(DFFf / 32, Dd / 32);
  k_castT<<<gT2, 256, 0, stream>>>(W1, w1T, Dd, DFFf);
  dim3 gT3(Dd / 32, DFFf / 32);
  k_castT<<<gT3, 256, 0, stream>>>(W2, w2T, DFFf, Dd);

  // 3) QKV projections: [4096,1024] x [1024,1024]
  dim3 gP(Dd / 128, (unsigned)(BS / 64));
  k_gemm<<<gP, 256, 0, stream>>>(xlnB, wqT, bq, nullptr, nullptr, qBf,
                                 (int)BS, Dd, Dd, 0, 0);
  k_gemm<<<gP, 256, 0, stream>>>(xlnB, wkT, bk, nullptr, nullptr, kBf,
                                 (int)BS, Dd, Dd, 0, 0);
  k_gemm<<<gP, 256, 0, stream>>>(xlnB, wvT, bv, nullptr, nullptr, vTf,
                                 (int)BS, Dd, Dd, 0, 1 /*V^T layout*/);

  // 4) attention
  dim3 gA(Ss / 128, Hh, Bb);
  k_attn<<<gA, 256, 0, stream>>>(qBf, kBf, vTf, attnB);

  // 5) output projection
  k_gemm<<<gP, 256, 0, stream>>>(attnB, woT, bo, nullptr, apF, nullptr,
                                 (int)BS, Dd, Dd, 0, 0);

  // 6) residual + LN2 + LN3
  k_post_attn<<<(int)BS, 256, 0, stream>>>(xlnF, apF, ln2_g, ln2_b, ln3_g,
                                           ln3_b, x2F, hBf);

  // 7) FFN up + GELU: [4096,1024] x [1024,4096]
  dim3 gU(DFFf / 128, (unsigned)(BS / 64));
  k_gemm<<<gU, 256, 0, stream>>>(hBf, w1T, b1, nullptr, nullptr, hhBf,
                                 (int)BS, DFFf, Dd, 1, 0);

  // 8) FFN down + residual -> d_out (fp32)
  k_gemm<<<gP, 256, 0, stream>>>(hhBf, w2T, b2, x2F, (float*)d_out, nullptr,
                                 (int)BS, Dd, DFFf, 0, 0);
}
